// GCN_66915590472397
// MI455X (gfx1250) — compile-verified, matrix operand order checked by reading, not run on hardware
//
#include <hip/hip_runtime.h>
#include <hip/hip_bf16.h>
#include <math.h>

// ---------------------------------------------------------------------------
// GCN forward for MI455X (gfx1250, wave32, WMMA).
// Dominant op: 13x dense A[2562x2562] @ H[2562x256] per batch of 8.
// A converted once to f16 (107MB -> L2-resident in the 192MB L2); all matmuls
// use v_wmma_f32_16x16x32_f16 with f32 accumulation. All B operands are kept
// column-major [N][K] so that每 fragment load is two ds_load_b128 (the 16-bit
// WMMA operand layout is two contiguous 8-element K runs per lane), and
// global->LDS staging stays 16B-contiguous for GLOBAL_LOAD_ASYNC_TO_LDS_B128
// (ASYNCcnt path). Epilogues fuse bias + A-term + residual + tanh.
// ---------------------------------------------------------------------------

typedef __attribute__((ext_vector_type(16))) _Float16 v16h;
typedef __attribute__((ext_vector_type(8)))  _Float16 v8h;
typedef __attribute__((ext_vector_type(4)))  _Float16 v4h;
typedef __attribute__((ext_vector_type(8)))  float    v8f;

#define B_SZ   8
#define N_PTS  2562
#define F_DIM  256
#define NP     2592        // N padded to a multiple of 32 (81 * 32)
#define IMGF   1280
#define C3_ 256
#define H3_ 56
#define W3_ 56
#define C4_ 512
#define H4_ 28
#define W4_ 28
#define C5_ 512
#define H5_ 14
#define W5_ 14

#define LKS 40             // LDS row stride (f16) for K-contiguous 32-wide rows

__device__ __forceinline__ v8f wmma_f16(v16h a, v16h b, v8f c) {
    return __builtin_amdgcn_wmma_f32_16x16x32_f16(
        /*neg_a=*/false, a, /*neg_b=*/false, b,
        /*c_mod=*/(short)0, c, /*reuse_a=*/false, /*reuse_b=*/false);
}

// Fragment load for the 16-bit 16x16x32 operand layout (ISA 7.12.2):
// lanes 0-15 hold K {0..7,16..23}, lanes 16-31 hold K {8..15,24..31}.
// rowBase points at the lane's K-contiguous 32-element row (16B aligned).
__device__ __forceinline__ v16h frag_ld(const _Float16* rowBase, int lane) {
    const _Float16* p = rowBase + ((lane < 16) ? 0 : 8);
    v8h lo = *(const v8h*)(p);        // K = base .. base+7
    v8h hi = *(const v8h*)(p + 16);   // K = base+16 .. base+23
    return __builtin_shufflevector(lo, hi, 0, 1, 2, 3, 4, 5, 6, 7,
                                           8, 9, 10, 11, 12, 13, 14, 15);
}

// --- 16-byte global -> LDS copy: async-to-LDS (ASYNCcnt) when available ----
#if __has_builtin(__builtin_amdgcn_global_load_async_to_lds_b128) && \
    __has_builtin(__builtin_amdgcn_s_wait_asynccnt)
#define HAS_ASYNC_LDS 1
#endif

typedef int vsi4 __attribute__((vector_size(4 * sizeof(int))));
#define AS1P __attribute__((address_space(1)))
#define AS3P __attribute__((address_space(3)))

__device__ __forceinline__ void cp16_g2l(const _Float16* __restrict__ g,
                                         _Float16* __restrict__ l) {
#if defined(HAS_ASYNC_LDS)
    __builtin_amdgcn_global_load_async_to_lds_b128(
        (AS1P vsi4*)(g), (AS3P vsi4*)(l), 0, 0);
#else
    *(float4*)l = *(const float4*)g;
#endif
}

__device__ __forceinline__ void async_join() {
#if defined(HAS_ASYNC_LDS)
    __builtin_amdgcn_s_wait_asynccnt(0);
#endif
}

// ---------------------------------------------------------------------------
// One-time conversions
// ---------------------------------------------------------------------------
__global__ void k_cvtA(const float* __restrict__ A, _Float16* __restrict__ A16) {
    long long idx = (long long)blockIdx.x * blockDim.x + threadIdx.x;
    const long long total = (long long)B_SZ * NP * NP;
    if (idx >= total) return;
    int b   = (int)(idx / ((long long)NP * NP));
    int rem = (int)(idx % ((long long)NP * NP));
    int r = rem / NP, c = rem % NP;
    float v = 0.f;
    if (r < N_PTS && c < N_PTS)
        v = A[((long long)b * N_PTS + r) * N_PTS + c];
    A16[idx] = (_Float16)v;
}

// f32 [K][Nc] row-major -> f16 [Nc][K] column-major (nmats matrices)
__global__ void k_cvtT(const float* __restrict__ in, _Float16* __restrict__ out,
                       int K, int Nc, int nmats) {
    long long idx = (long long)blockIdx.x * blockDim.x + threadIdx.x;
    long long per = (long long)K * Nc;
    if (idx >= per * nmats) return;
    int mat = (int)(idx / per);
    int rem = (int)(idx % per);
    int n = rem / K, k = rem % K;
    out[idx] = (_Float16)in[(long long)mat * per + (long long)k * Nc + n];
}

// ---------------------------------------------------------------------------
// Bilinear feature gather + c_f + s_f (K=1280 WMMA) -> cs16 [B*NP, 512] f16
// cols [0,256) = c_f, cols [256,512) = s_f.
// ---------------------------------------------------------------------------
__device__ __forceinline__ float bilin(const float* __restrict__ base,
                                       int H, int W, int ch, float x, float y) {
    float x1 = floorf(x), x2 = ceilf(x);
    float y1 = floorf(y), y2 = ceilf(y);
    int xi1 = (int)x1, xi2 = (int)x2, yi1 = (int)y1, yi2 = (int)y2;
    const float* f = base + (long long)ch * H * W;
    float v11 = f[yi1 * W + xi1];
    float v12 = f[yi2 * W + xi1];
    float v21 = f[yi1 * W + xi2];
    float v22 = f[yi2 * W + xi2];
    return v11 * (x2 - x) * (y2 - y) + v12 * (x2 - x) * (y - y1) +
           v21 * (x - x1) * (y2 - y) + v22 * (x - x1) * (y - y1);
}

#define SLD (IMGF + 8)

__global__ __launch_bounds__(256) void k_point(
    const float* __restrict__ c_prev, const float* __restrict__ conv3,
    const float* __restrict__ conv4, const float* __restrict__ conv5,
    const float* __restrict__ Wpc, const float* __restrict__ bpc,
    const _Float16* __restrict__ WpsT, const float* __restrict__ bps,
    _Float16* __restrict__ cs16)
{
    __shared__ alignas(16) _Float16 sprev[16 * SLD];   // 16 x 1280 f16 tile
    __shared__ alignas(16) _Float16 sWt[256 * LKS];    // 256 n-rows x 32 k
    __shared__ float sc[16][3];
    const int tid = threadIdx.x;
    const int tileM = blockIdx.x, b = blockIdx.y;

    if (tid < 48) {
        int nl = tid / 3, d = tid % 3;
        int node = tileM * 16 + nl;
        sc[nl][d] = (node < N_PTS)
                  ? c_prev[((long long)b * N_PTS + node) * 3 + d] : 0.f;
    }
    __syncthreads();

    // gather bilinear features (faithful to reference: conv3 coords scaled to
    // pixels, conv4/conv5 sampled with coords in [0,1])
    for (int idx = tid; idx < 16 * IMGF; idx += 256) {
        int nl = idx / IMGF, ch = idx % IMGF;
        int node = tileM * 16 + nl;
        float val = 0.f;
        if (node < N_PTS) {
            float sx = (sc[nl][0] + 1.f) * 0.5f;
            float sy = (sc[nl][1] + 1.f) * 0.5f;
            if (ch < C3_) {
                val = bilin(conv3 + (long long)b * C3_ * H3_ * W3_, H3_, W3_,
                            ch, sx * (W3_ - 1), sy * (H3_ - 1));
            } else if (ch < C3_ + C4_) {
                val = bilin(conv4 + (long long)b * C4_ * H4_ * W4_, H4_, W4_,
                            ch - C3_, sx, sy);
            } else {
                val = bilin(conv5 + (long long)b * C5_ * H5_ * W5_, H5_, W5_,
                            ch - C3_ - C4_, sx, sy);
            }
        }
        sprev[nl * SLD + ch] = (_Float16)val;
    }

    // c_f = tanh(c_prev @ Wpc + bpc) (K = 3, pure VALU) -> cs16 cols [0,256)
    for (int idx = tid; idx < 16 * F_DIM; idx += 256) {
        int nl = idx >> 8, f = idx & 255;
        int node = tileM * 16 + nl;
        float v = 0.f;
        if (node < N_PTS)
            v = tanhf(sc[nl][0] * Wpc[f] + sc[nl][1] * Wpc[F_DIM + f] +
                      sc[nl][2] * Wpc[2 * F_DIM + f] + bpc[f]);
        cs16[((long long)b * NP + tileM * 16 + nl) * 512 + f] = (_Float16)v;
    }
    __syncthreads();

    // s_f = tanh(sprev @ Wps + bps), 16x256 out, K=1280. 8 waves x 2 subtiles.
    const int wave = tid >> 5, lane = tid & 31;
    v8f acc0 = {}, acc1 = {};
    for (int k0 = 0; k0 < IMGF; k0 += 32) {
        // stage 256x32 column-major weight tile (16B K-contiguous chunks)
        for (int idx = tid; idx < 1024; idx += 256) {
            int n = idx >> 2, c = idx & 3;
            cp16_g2l(WpsT + (long long)n * IMGF + k0 + c * 8,
                     sWt + n * LKS + c * 8);
        }
        async_join();
        __syncthreads();
        v16h a = frag_ld(sprev + (lane & 15) * SLD + k0, lane);
        int n0 = (wave * 2 + 0) * 16 + (lane & 15);
        int n1 = (wave * 2 + 1) * 16 + (lane & 15);
        v16h bf0 = frag_ld(sWt + n0 * LKS, lane);
        v16h bf1 = frag_ld(sWt + n1 * LKS, lane);
        acc0 = wmma_f16(a, bf0, acc0);
        acc1 = wmma_f16(a, bf1, acc1);
        __syncthreads();
    }
    #pragma unroll
    for (int t = 0; t < 2; t++) {
        v8f acc = t ? acc1 : acc0;
        int n = (wave * 2 + t) * 16 + (lane & 15);
        #pragma unroll
        for (int r = 0; r < 8; r++) {
            int m = r + ((lane >= 16) ? 8 : 0);
            float v = tanhf(acc[r] + bps[n]);
            cs16[((long long)b * NP + tileM * 16 + m) * 512 + 256 + n] =
                (_Float16)v;
        }
    }
}

// ---------------------------------------------------------------------------
// Fill xf cols [256,512) with x_prev (zero for padded nodes)
// ---------------------------------------------------------------------------
__global__ void k_xprev(const float* __restrict__ x_prev, float* __restrict__ xf) {
    long long idx = (long long)blockIdx.x * blockDim.x + threadIdx.x;
    const long long total = (long long)B_SZ * NP * F_DIM;
    if (idx >= total) return;
    int f = (int)(idx & 255);
    long long row = idx >> 8;
    int n = (int)(row % NP);
    float v = (n < N_PTS)
            ? x_prev[(((row / NP) * (long long)N_PTS) + n) * F_DIM + f] : 0.f;
    xf[row * 512 + 256 + f] = v;
}

// ---------------------------------------------------------------------------
// Generic WMMA GEMM per batch (grid = (NP/16, 1, B)):
//   out = act( X[NP,K] @ B^T + bias + zT + res )
// X: f32 (converted while staging) or f16 (16B async chunks).
// Bt: f16 column-major [256][K]. zT: optional f32 [256][NP] (transposed).
// outF: f32, row-major (ldo) or transposed [256][NP] (outFT).
// outHT: optional f16 transposed [256][NP] mirror (einsum B operand).
// ---------------------------------------------------------------------------
__global__ __launch_bounds__(128) void k_gemm(
    const void* __restrict__ Xv, int xIsF16, long long xStride, int ldx,
    const _Float16* __restrict__ Bt, long long bStride, int ldb,
    const float* __restrict__ bias,
    const float* __restrict__ zT, long long zStride,
    const float* __restrict__ res, long long resStride, int ldr,
    float* __restrict__ outF, long long outFStride, int ldo, int outFT,
    _Float16* __restrict__ outHT, long long outHStride, int ldh,
    int K, int act)
{
    __shared__ alignas(16) _Float16 lX[16 * LKS];    // 16 rows x 32 k
    __shared__ alignas(16) _Float16 lWt[256 * LKS];  // 256 n-rows x 32 k
    const int tid = threadIdx.x, wave = tid >> 5, lane = tid & 31;
    const int m0 = blockIdx.x * 16;
    const int z = blockIdx.z;
    const _Float16* Xh = (const _Float16*)Xv + z * xStride;
    const float*    Xf = (const float*)Xv + z * xStride;
    const _Float16* Bz = Bt + z * bStride;

    v8f acc[4] = {{}, {}, {}, {}};
    for (int k0 = 0; k0 < K; k0 += 32) {
        // ---- stage B tile: 256 n-rows x 64B, K-contiguous 16B chunks ----
        for (int idx = tid; idx < 1024; idx += 128) {
            int n = idx >> 2, c = idx & 3;
            cp16_g2l(Bz + (long long)n * ldb + k0 + c * 8,
                     lWt + n * LKS + c * 8);
        }
        // ---- stage A tile: 16 rows x 32 k ----
        if (xIsF16) {
            if (tid < 64) {
                int r = tid >> 2, c = tid & 3;
                cp16_g2l(Xh + (long long)(m0 + r) * ldx + k0 + c * 8,
                         lX + r * LKS + c * 8);
            }
        } else {
            // 128 threads, one float4 each: coalesced b128 read + cvt + b64 store
            int m = tid >> 3, c4 = (tid & 7) * 4;
            float4 xv = *(const float4*)(Xf + (long long)(m0 + m) * ldx + k0 + c4);
            v4h hx;
            hx[0] = (_Float16)xv.x; hx[1] = (_Float16)xv.y;
            hx[2] = (_Float16)xv.z; hx[3] = (_Float16)xv.w;
            *(v4h*)(lX + m * LKS + c4) = hx;
        }
        // warm next K-chunk of a few B rows (global_prefetch_b8)
        if (k0 + 32 < K && tid < 32)
            __builtin_prefetch(Bz + (long long)(tid * 8) * ldb + k0 + 32, 0, 1);
        async_join();
        __syncthreads();

        v16h a = frag_ld(lX + (lane & 15) * LKS, lane);
        #pragma unroll
        for (int t = 0; t < 4; t++) {
            int n = (wave * 4 + t) * 16 + (lane & 15);
            v16h bfrag = frag_ld(lWt + n * LKS, lane);
            acc[t] = wmma_f16(a, bfrag, acc[t]);
        }
        __syncthreads();
    }

    const int baseRow = m0 + ((lane >= 16) ? 8 : 0);
    #pragma unroll
    for (int t = 0; t < 4; t++) {
        int n = (wave * 4 + t) * 16 + (lane & 15);
        float vals[8];
        const float* zcol = zT ? zT + z * zStride + (long long)n * NP + baseRow
                               : nullptr;
        const float* rrow = res ? res + z * resStride : nullptr;
        #pragma unroll
        for (int r = 0; r < 8; r++) {
            float v = acc[t][r];
            if (bias) v += bias[n];
            if (zcol) v += zcol[r];                       // 2x float4 loads
            if (rrow) v += rrow[(long long)(baseRow + r) * ldr + n];
            if (act)  v = tanhf(v);
            vals[r] = v;
        }
        if (outF) {
            float* o = outF + z * outFStride;
            if (outFT) {                                   // transposed: 2x float4
                float4* p = (float4*)(o + (long long)n * ldo + baseRow);
                p[0] = make_float4(vals[0], vals[1], vals[2], vals[3]);
                p[1] = make_float4(vals[4], vals[5], vals[6], vals[7]);
            } else {
                #pragma unroll
                for (int r = 0; r < 8; r++)
                    o[(long long)(baseRow + r) * ldo + n] = vals[r];
            }
        }
        if (outHT) {                                       // one v8h store
            v8h hv;
            #pragma unroll
            for (int r = 0; r < 8; r++) hv[r] = (_Float16)vals[r];
            *(v8h*)(outHT + z * outHStride + (long long)n * ldh + baseRow) = hv;
        }
    }
}

// ---------------------------------------------------------------------------
// Outputs: x -> d_out[0 .. B*N*F), c = tanh(x @ Wf + bf) -> tail
// ---------------------------------------------------------------------------
__global__ void k_out_x(const float* __restrict__ xf, float* __restrict__ out) {
    long long idx = (long long)blockIdx.x * blockDim.x + threadIdx.x;
    const long long total = (long long)B_SZ * N_PTS * F_DIM;
    if (idx >= total) return;
    int f = (int)(idx & 255);
    long long bn = idx >> 8;
    int n = (int)(bn % N_PTS);
    int b = (int)(bn / N_PTS);
    out[idx] = xf[((long long)b * NP + n) * 512 + f];
}

__global__ void k_out_c(const float* __restrict__ xf,
                        const float* __restrict__ Wf,
                        const float* __restrict__ bf,
                        float* __restrict__ out) {
    int idx = blockIdx.x * blockDim.x + threadIdx.x;
    if (idx >= B_SZ * N_PTS * 3) return;
    int d = idx % 3;
    int bn = idx / 3;
    int n = bn % N_PTS, b = bn / N_PTS;
    const float* xr = xf + ((long long)b * NP + n) * 512;
    float s = bf[d];
    for (int k = 0; k < F_DIM; k++) s += xr[k] * Wf[k * 3 + d];
    out[idx] = tanhf(s);
}

// ---------------------------------------------------------------------------
extern "C" void kernel_launch(void* const* d_in, const int* in_sizes, int n_in,
                              void* d_out, int out_size, void* d_ws, size_t ws_size,
                              hipStream_t stream) {
    const float* x_prev = (const float*)d_in[0];
    const float* c_prev = (const float*)d_in[1];
    const float* A      = (const float*)d_in[2];
    const float* conv3  = (const float*)d_in[3];
    const float* conv4  = (const float*)d_in[4];
    const float* conv5  = (const float*)d_in[5];
    const float* W0  = (const float*)d_in[6];
    const float* b0  = (const float*)d_in[7];
    const float* W1  = (const float*)d_in[8];
    const float* b1  = (const float*)d_in[9];
    const float* Wg  = (const float*)d_in[10];
    const float* bg  = (const float*)d_in[11];
    const float* Wpc = (const float*)d_in[12];
    const float* bpc = (const float*)d_in[13];
    const float* Wps = (const float*)d_in[14];
    const float* bps = (const float*)d_in[15];
    const float* Wp  = (const float*)d_in[16];
    const float* bp  = (const float*)d_in[17];
    const float* Wf  = (const float*)d_in[18];
    const float* bf  = (const float*)d_in[19];
    (void)in_sizes; (void)n_in; (void)out_size; (void)ws_size;

    char* ws = (char*)d_ws;
    size_t off = 0;
    auto alloc = [&](size_t bytes) -> void* {
        void* p = ws + off;
        off = (off + bytes + 255) & ~(size_t)255;
        return p;
    };
    _Float16* A16   = (_Float16*)alloc((size_t)B_SZ * NP * NP * 2);   // ~107 MB
    _Float16* cs16  = (_Float16*)alloc((size_t)B_SZ * NP * 512 * 2);  // concat(c_f,s_f)
    _Float16* h16t  = (_Float16*)alloc((size_t)B_SZ * 256 * NP * 2);  // [B][256][NP]
    _Float16* WpsT  = (_Float16*)alloc((size_t)256 * IMGF * 2);
    _Float16* WpT   = (_Float16*)alloc((size_t)256 * 512 * 2);
    _Float16* W0T   = (_Float16*)alloc((size_t)256 * 512 * 2);
    _Float16* W1T   = (_Float16*)alloc((size_t)256 * 512 * 2);
    _Float16* WgT   = (_Float16*)alloc((size_t)24 * 256 * 256 * 2);
    float*    xf    = (float*)alloc((size_t)B_SZ * NP * 512 * 4);     // x (ld 512)
    float*    zfT   = (float*)alloc((size_t)B_SZ * 256 * NP * 4);     // (A@h)^T

    // --- one-time precision conversion (A stays L2-resident in f16) ---
    const long long totA = (long long)B_SZ * NP * NP;
    k_cvtA<<<(unsigned)((totA + 255) / 256), 256, 0, stream>>>(A, A16);
    k_cvtT<<<(IMGF * 256 + 255) / 256, 256, 0, stream>>>(Wps, WpsT, IMGF, 256, 1);
    k_cvtT<<<(512 * 256 + 255) / 256, 256, 0, stream>>>(Wp, WpT, 512, 256, 1);
    k_cvtT<<<(512 * 256 + 255) / 256, 256, 0, stream>>>(W0, W0T, 512, 256, 1);
    k_cvtT<<<(512 * 256 + 255) / 256, 256, 0, stream>>>(W1, W1T, 512, 256, 1);
    k_cvtT<<<(24 * 256 * 256 + 255) / 256, 256, 0, stream>>>(Wg, WgT, 256, 256, 24);

    // --- point features: c_f + bilinear gather + s_f (WMMA K=1280) ---
    dim3 gp(NP / 16, B_SZ);
    k_point<<<gp, 256, 0, stream>>>(c_prev, conv3, conv4, conv5,
                                    Wpc, bpc, WpsT, bps, cs16);

    const long long totXP = (long long)B_SZ * NP * 256;
    k_xprev<<<(unsigned)((totXP + 255) / 256), 256, 0, stream>>>(x_prev, xf);

    dim3 gE(NP / 16, 1, B_SZ);
    const long long sX512 = (long long)NP * 512;   // xf / cs16 batch stride
    const long long sA    = (long long)NP * NP;    // A16 batch stride
    const long long sT    = (long long)256 * NP;   // h16t / zfT batch stride

    // ffs = tanh(concat(c_f, s_f) @ Wp + bp) -> xf cols [0,256)
    k_gemm<<<gE, 128, 0, stream>>>(cs16, 1, sX512, 512, WpT, 0, 512, bp,
                                   nullptr, 0, nullptr, 0, 0,
                                   xf, sX512, 512, 0, nullptr, 0, 0, 512, 1);

    // layer 0: x = tanh(x@W0 + b0 + A @ (x@W1 + b1)),  x is 512-wide here
    k_gemm<<<gE, 128, 0, stream>>>(xf, 0, sX512, 512, W1T, 0, 512, b1,
                                   nullptr, 0, nullptr, 0, 0,
                                   nullptr, 0, 0, 0, h16t, sT, NP, 512, 0);
    k_gemm<<<gE, 128, 0, stream>>>(A16, 1, sA, NP, h16t, sT, NP, nullptr,
                                   nullptr, 0, nullptr, 0, 0,
                                   zfT, sT, NP, 1, nullptr, 0, 0, NP, 0);
    k_gemm<<<gE, 128, 0, stream>>>(xf, 0, sX512, 512, W0T, 0, 512, b0,
                                   zfT, sT, nullptr, 0, 0,
                                   xf, sX512, 512, 0, nullptr, 0, 0, 512, 1);

    // 12 residual GCN layers
    for (int k = 0; k < 12; k++) {
        const _Float16* WnT = WgT + (size_t)(2 * k + 1) * 65536;
        const _Float16* WsT = WgT + (size_t)(2 * k) * 65536;
        const float* bn = bg + (2 * k + 1) * 256;
        const float* bs = bg + (2 * k) * 256;
        k_gemm<<<gE, 128, 0, stream>>>(xf, 0, sX512, 512, WnT, 0, 256, bn,
                                       nullptr, 0, nullptr, 0, 0,
                                       nullptr, 0, 0, 0, h16t, sT, NP, 256, 0);
        k_gemm<<<gE, 128, 0, stream>>>(A16, 1, sA, NP, h16t, sT, NP, nullptr,
                                       nullptr, 0, nullptr, 0, 0,
                                       zfT, sT, NP, 1, nullptr, 0, 0, NP, 0);
        k_gemm<<<gE, 128, 0, stream>>>(xf, 0, sX512, 512, WsT, 0, 256, bs,
                                       zfT, sT, xf, sX512, 512,
                                       xf, sX512, 512, 0, nullptr, 0, 0, 256, 1);
    }

    // outputs: x then c (tuple order)
    const long long totX = (long long)B_SZ * N_PTS * F_DIM;
    k_out_x<<<(unsigned)((totX + 255) / 256), 256, 0, stream>>>(xf, (float*)d_out);
    const int totC = B_SZ * N_PTS * 3;
    k_out_c<<<(totC + 255) / 256, 256, 0, stream>>>(xf, Wf, bf,
                                                    (float*)d_out + totX);
}
